// SpectralVSAArchive_9234179686585
// MI455X (gfx1250) — matrix operand order repaired and checked by reading, not compiled, not run on hardware
//
#include <hip/hip_runtime.h>
#include <hip/hip_bf16.h>
#include <math.h>

// ---------------------------------------------------------------------------
// Types for CDNA5 WMMA (wave32): v16bf = 16x bf16 (8 VGPRs), v8f = f32 accum.
// ---------------------------------------------------------------------------
typedef __bf16 bf16_t;
typedef __attribute__((ext_vector_type(16))) __bf16 v16bf;
typedef __attribute__((ext_vector_type(8)))  float  v8f;

#define D_MODEL 1024
#define KBANDS  32
#define WWIN    256
#define NWIN    256
#define NRB     8
#define ALPHA   0.9f

// Native f32->bf16 (RNE) -- lowers to v_cvt_pk_bf16_f32 class ops on gfx1250.
__device__ __forceinline__ bf16_t f2bf(float f) { return (bf16_t)f; }
__device__ __forceinline__ float bf2f(bf16_t h) {
    unsigned short s = __builtin_bit_cast(unsigned short, h);
    return __builtin_bit_cast(float, ((unsigned)s) << 16);
}
__device__ __forceinline__ float sigmoidf_(float x) { return 1.0f / (1.0f + expf(-x)); }

// Low 32 bits of a generic pointer to __shared__ data == LDS byte offset
// (generic LDS addr = {SHARED_BASE[63:32], lds_offset[31:0]}, ISA 10.2).
__device__ __forceinline__ unsigned lds_addr_u32(const void* p) {
    return (unsigned)(size_t)p;
}

// GLOBAL_LOAD_ASYNC_TO_LDS_B128: copy 16B/lane global -> LDS, ASYNCcnt-tracked.
__device__ __forceinline__ void async_copy_b128(unsigned lds_byte_addr,
                                                const void* gaddr) {
    asm volatile("global_load_async_to_lds_b128 %0, %1, off"
                 :: "v"(lds_byte_addr), "v"(gaddr) : "memory");
}
__device__ __forceinline__ void wait_asynccnt0() {
    asm volatile("s_wait_asynccnt 0x0" ::: "memory");
}

__device__ __forceinline__ v8f wmma_bf16(v16bf a, v16bf b, v8f c) {
    return __builtin_amdgcn_wmma_f32_16x16x32_bf16(false, a, false, b,
                                                   (short)0, c, false, false);
}

// A fragment (16x32 bf16) with k = 16*(lane/16) + s, LDS tile [M][K] row-major, stride in
// bf16 elems (kept even so K-pairs are DWORD aligned -> ds_load_2addr_b32).
__device__ __forceinline__ v16bf load_a_frag(const bf16_t* lds, int stride,
                                             int m0, int k0, int lane) {
    const int r  = m0 + (lane & 15);
    const int kb = (lane < 16) ? 0 : 16;
    const bf16_t* row = lds + r * stride + k0;
    v16bf a;
#pragma unroll
    for (int p = 0; p < 4; ++p) {
        a[2 * p]         = row[kb + 2 * p];
        a[2 * p + 1]     = row[kb + 2 * p + 1];
        a[8 + 2 * p]     = row[kb + 8 + 2 * p];
        a[8 + 2 * p + 1] = row[kb + 8 + 2 * p + 1];
    }
    return a;
}

// B fragment (32x16 bf16) from an LDS tile stored [N][K] row-major (stride 40
// bf16 = 80B rows): each lane reads 32 contiguous, 16B-aligned bytes
// -> 2x ds_load_b128, conflict-free (20 DW stride cycles all 64 banks).
__device__ __forceinline__ v16bf load_b_frag_nk(const bf16_t* lds, int stride,
                                                int n0, int k0, int lane) {
    const bf16_t* p = lds + (n0 + (lane & 15)) * stride + k0
                          + ((lane < 16) ? 0 : 16);
    struct P { uint4 a, b; } t;
    t.a = *reinterpret_cast<const uint4*>(p);
    t.b = *reinterpret_cast<const uint4*>(p + 8);
    return __builtin_bit_cast(v16bf, t);
}

// B fragment from an f32 LDS tile stored [K][N] (async-staged x), converting
// to bf16 at fragment-assembly time (native packed cvt).
__device__ __forceinline__ v16bf load_b_frag_f32kn(const float* lds, int strideDW,
                                                   int k0, int n0, int lane) {
    const int c  = n0 + (lane & 15);
    const int kb = k0 + ((lane < 16) ? 0 : 16);
    v16bf b;
#pragma unroll
    for (int i = 0; i < 16; ++i) b[i] = f2bf(lds[(kb + i) * strideDW + c]);
    return b;
}

// ---------------------------------------------------------------------------
// Kernel 1: y[b,k,d] = sum_n cheby[k,n] * x[b,n,d]   (per-window 32x1024 GEMM)
// x chunks staged f32 via GLOBAL_LOAD_ASYNC_TO_LDS_B128 (pure byte copy).
// grid (4 d-tiles of 256, 256 windows), block 256 (8 waves).
// ---------------------------------------------------------------------------
__global__ __launch_bounds__(256)
void k1_cheby_project(const float* __restrict__ x,
                      const float* __restrict__ cheby,
                      bf16_t* __restrict__ y_ws) {
    __shared__ __align__(16) bf16_t sA[32 * 258];  // cheby bf16 (129 DW rows, odd)
    __shared__ __align__(16) float  sX[32 * 260];  // x chunk f32 (1040B rows, 16B mult)

    const int tid  = threadIdx.x;
    const int lane = tid & 31;
    const int w    = tid >> 5;
    const int b    = blockIdx.y;
    const int d0   = blockIdx.x * 256;

    for (int idx = tid; idx < 32 * 256; idx += 256) {
        const int k = idx >> 8, n = idx & 255;
        sA[k * 258 + n] = f2bf(cheby[idx]);
    }

    v8f acc[2][2];
#pragma unroll
    for (int mi = 0; mi < 2; ++mi)
#pragma unroll
        for (int ni = 0; ni < 2; ++ni)
            acc[mi][ni] = v8f{0.f, 0.f, 0.f, 0.f, 0.f, 0.f, 0.f, 0.f};

    const float* xb = x + (size_t)b * WWIN * D_MODEL;

    for (int n0 = 0; n0 < WWIN; n0 += 32) {
        __syncthreads();                 // prior consumers done before overwrite
        // async-stage 32 rows x 1KB: 2048 x B128 blocks, 8 per thread
        for (int t = tid; t < 2048; t += 256) {
            const int r = t >> 6, blk = t & 63;
            async_copy_b128(lds_addr_u32(sX + r * 260 + blk * 4),
                            xb + (size_t)(n0 + r) * D_MODEL + d0 + blk * 4);
        }
        wait_asynccnt0();
        __syncthreads();
        const v16bf a0 = load_a_frag(sA, 258, 0, n0, lane);
        const v16bf a1 = load_a_frag(sA, 258, 16, n0, lane);
#pragma unroll
        for (int ni = 0; ni < 2; ++ni) {
            const v16bf bf = load_b_frag_f32kn(sX, 260, 0, (2 * w + ni) * 16, lane);
            acc[0][ni] = wmma_bf16(a0, bf, acc[0][ni]);
            acc[1][ni] = wmma_bf16(a1, bf, acc[1][ni]);
        }
    }

    const int cl    = lane & 15;
    const int rbase = (lane < 16) ? 0 : 8;
#pragma unroll
    for (int mi = 0; mi < 2; ++mi)
#pragma unroll
        for (int ni = 0; ni < 2; ++ni)
#pragma unroll
            for (int v = 0; v < 8; ++v) {
                const int kk = mi * 16 + rbase + v;
                const int dg = d0 + (2 * w + ni) * 16 + cl;
                y_ws[((size_t)b * KBANDS + kk) * D_MODEL + dg] = f2bf(acc[mi][ni][v]);
            }
}

// ---------------------------------------------------------------------------
// Kernel 2: z = (1-gc)*y + gc*(y @ Wc^T); band scale; complex role bind;
// EMA blend; trust-weighted unbind -> retr_pre[b,d].
// Double-buffered pipeline: A (y, bf16) via async-to-LDS issued ahead of the
// WMMAs; B (compress_W, natural [e][d] layout) via register pipeline.
// grid (8 e-tiles of 128, 256 windows), block 256 (wave w owns 16 e-cols).
// ---------------------------------------------------------------------------
__global__ __launch_bounds__(256)
void k2_compress_bind(const bf16_t* __restrict__ y_ws,
                      const float* __restrict__ compress_W,
                      const float* __restrict__ compress_gate,
                      const float* __restrict__ w_bands,
                      const float* __restrict__ freq_bias,
                      const float* __restrict__ lanczos,
                      const float* __restrict__ band_trust,
                      const float* __restrict__ roles_r,
                      const float* __restrict__ roles_i,
                      const float* __restrict__ vmem_r,
                      const float* __restrict__ vmem_i,
                      float* __restrict__ retr_pre) {
    __shared__ __align__(16) bf16_t sA[2][32 * 40];    // y chunk  [k][d], 80B rows
    __shared__ __align__(16) bf16_t sB[2][128 * 40];   // Wc chunk [e][d], 80B rows

    const int tid  = threadIdx.x;
    const int lane = tid & 31;
    const int w    = tid >> 5;
    const int b    = blockIdx.y;
    const int e0   = blockIdx.x * 128;

    const bf16_t* yb = y_ws + (size_t)b * KBANDS * D_MODEL;

    float rb[16];
    auto asyncA = [&](int d0, int p) {   // 32 rows x 64B = 128 B128 copies
        if (tid < 128) {
            const int r = tid >> 2, blk = tid & 3;
            async_copy_b128(lds_addr_u32(sA[p] + r * 40 + blk * 8),
                            yb + (size_t)r * D_MODEL + d0 + blk * 8);
        }
    };
    auto loadB = [&](int d0) {
#pragma unroll
        for (int u = 0; u < 16; ++u) {
            const int idx = tid + u * 256;
            const int j = idx >> 5, i = idx & 31;
            rb[u] = compress_W[(size_t)(e0 + j) * D_MODEL + d0 + i];
        }
    };
    auto storeB = [&](int p) {
#pragma unroll
        for (int u = 0; u < 16; ++u) {
            const int idx = tid + u * 256;
            const int j = idx >> 5, i = idx & 31;
            sB[p][j * 40 + i] = f2bf(rb[u]);
        }
    };

    v8f acc[2];
    acc[0] = v8f{0.f, 0.f, 0.f, 0.f, 0.f, 0.f, 0.f, 0.f};
    acc[1] = acc[0];

    loadB(0);
    asyncA(0, 0);
    storeB(0);
    wait_asynccnt0();
    __syncthreads();

    int p = 0;
    for (int d0 = 0; d0 < D_MODEL; d0 += 32) {
        const bool more = (d0 + 32) < D_MODEL;
        if (more) { asyncA(d0 + 32, p ^ 1); loadB(d0 + 32); }   // prefetch next
        const v16bf a0 = load_a_frag(sA[p], 40, 0, 0, lane);
        const v16bf a1 = load_a_frag(sA[p], 40, 16, 0, lane);
        const v16bf bf = load_b_frag_nk(sB[p], 40, w * 16, 0, lane);
        acc[0] = wmma_bf16(a0, bf, acc[0]);
        acc[1] = wmma_bf16(a1, bf, acc[1]);
        if (more) storeB(p ^ 1);
        wait_asynccnt0();
        __syncthreads();
        p ^= 1;
    }

    // Epilogue: each lane holds 16 k-rows (two m-tiles x 8) of one column dg.
    const float gc  = sigmoidf_(compress_gate[0]);
    const int cl    = lane & 15;
    const int rbase = (lane < 16) ? 0 : 8;
    const int dg    = e0 + w * 16 + cl;

    float vr = 0.f, vi = 0.f;
#pragma unroll
    for (int mi = 0; mi < 2; ++mi)
#pragma unroll
        for (int v = 0; v < 8; ++v) {
            const int kk = mi * 16 + rbase + v;
            const float yv = bf2f(yb[(size_t)kk * D_MODEL + dg]);
            float z = gc * acc[mi][v] + (1.0f - gc) * yv;
            const float sc = (kk == 0) ? (1.0f / (float)WWIN) : (2.0f / (float)WWIN);
            z *= sc * lanczos[kk] * (w_bands[kk] + freq_bias[kk]);
            vr += z * roles_r[kk * D_MODEL + dg];
            vi += z * roles_i[kk * D_MODEL + dg];
        }
    vr += __shfl_xor(vr, 16, 32);        // combine k-half lanes (same column)
    vi += __shfl_xor(vi, 16, 32);

    const float Vr = ALPHA * vmem_r[dg] + (1.0f - ALPHA) * vr;
    const float Vi = ALPHA * vmem_i[dg] + (1.0f - ALPHA) * vi;

    float Sr = 0.f, Si = 0.f;
#pragma unroll
    for (int kk = 0; kk < NRB; ++kk) {
        const float t = sigmoidf_(band_trust[kk]);
        Sr += t * roles_r[kk * D_MODEL + dg];
        Si += t * roles_i[kk * D_MODEL + dg];
    }
    if (lane < 16)
        retr_pre[(size_t)b * D_MODEL + dg] = Vr * Sr + Vi * Si;
}

// ---------------------------------------------------------------------------
// Kernel 3: retr_g[b,e] = g * sum_d retr_pre[b,d] * retrieve_W[e,d]
// Double-buffered register pipeline for both A and B stages.
// grid (8 e-tiles of 128, 4 b-tiles of 64), block 256 (8 waves).
// ---------------------------------------------------------------------------
__global__ __launch_bounds__(256)
void k3_retrieve(const float* __restrict__ retr_pre,
                 const float* __restrict__ retrieve_W,
                 const float* __restrict__ inject_gate,
                 const float* __restrict__ blend_gate,
                 float* __restrict__ retr_g) {
    __shared__ __align__(16) bf16_t sA[2][64 * 40];    // retr_pre chunk [b][d]
    __shared__ __align__(16) bf16_t sB[2][128 * 40];   // Wr chunk      [e][d]

    const int tid  = threadIdx.x;
    const int lane = tid & 31;
    const int w    = tid >> 5;
    const int b0   = blockIdx.y * 64;
    const int e0   = blockIdx.x * 128;

    float ra[8], rb[16];
    auto loadA = [&](int d0) {
#pragma unroll
        for (int u = 0; u < 8; ++u) {
            const int idx = tid + u * 256;
            const int r = idx >> 5, c = idx & 31;
            ra[u] = retr_pre[(size_t)(b0 + r) * D_MODEL + d0 + c];
        }
    };
    auto storeA = [&](int p) {
#pragma unroll
        for (int u = 0; u < 8; ++u) {
            const int idx = tid + u * 256;
            const int r = idx >> 5, c = idx & 31;
            sA[p][r * 40 + c] = f2bf(ra[u]);
        }
    };
    auto loadB = [&](int d0) {
#pragma unroll
        for (int u = 0; u < 16; ++u) {
            const int idx = tid + u * 256;
            const int j = idx >> 5, i = idx & 31;
            rb[u] = retrieve_W[(size_t)(e0 + j) * D_MODEL + d0 + i];
        }
    };
    auto storeB = [&](int p) {
#pragma unroll
        for (int u = 0; u < 16; ++u) {
            const int idx = tid + u * 256;
            const int j = idx >> 5, i = idx & 31;
            sB[p][j * 40 + i] = f2bf(rb[u]);
        }
    };

    v8f acc[4];
#pragma unroll
    for (int mi = 0; mi < 4; ++mi)
        acc[mi] = v8f{0.f, 0.f, 0.f, 0.f, 0.f, 0.f, 0.f, 0.f};

    loadA(0); loadB(0);
    storeA(0); storeB(0);
    __syncthreads();

    int p = 0;
    for (int d0 = 0; d0 < D_MODEL; d0 += 32) {
        const bool more = (d0 + 32) < D_MODEL;
        if (more) { loadA(d0 + 32); loadB(d0 + 32); }   // issue next-chunk loads
        const v16bf bf = load_b_frag_nk(sB[p], 40, w * 16, 0, lane);
#pragma unroll
        for (int mi = 0; mi < 4; ++mi) {
            const v16bf a = load_a_frag(sA[p], 40, mi * 16, 0, lane);
            acc[mi] = wmma_bf16(a, bf, acc[mi]);
        }
        if (more) { storeA(p ^ 1); storeB(p ^ 1); }
        __syncthreads();
        p ^= 1;
    }

    const float g   = sigmoidf_(inject_gate[0]) * blend_gate[0];
    const int cl    = lane & 15;
    const int rbase = (lane < 16) ? 0 : 8;
    const int eg    = e0 + w * 16 + cl;
#pragma unroll
    for (int mi = 0; mi < 4; ++mi)
#pragma unroll
        for (int v = 0; v < 8; ++v)
            retr_g[(size_t)(b0 + mi * 16 + rbase + v) * D_MODEL + eg] = g * acc[mi][v];
}

// ---------------------------------------------------------------------------
// Kernel 4: out[b,w,e] = x[b,w,e] + retr_g[b,e]   (pure streaming, float4)
// ---------------------------------------------------------------------------
__global__ __launch_bounds__(256)
void k4_inject(const float4* __restrict__ x4,
               const float* __restrict__ retr_g,
               float4* __restrict__ out4) {
    const size_t total  = (size_t)NWIN * WWIN * (D_MODEL / 4);   // 16,777,216
    const size_t stride = (size_t)gridDim.x * blockDim.x;
    for (size_t i = (size_t)blockIdx.x * blockDim.x + threadIdx.x; i < total; i += stride) {
        const size_t b  = i >> 16;          // 65536 float4 per window
        const size_t e4 = i & 255;          // 256 float4 per row
        const float4 xv = x4[i];
        const float4 rv = reinterpret_cast<const float4*>(retr_g)[b * 256 + e4];
        float4 o;
        o.x = xv.x + rv.x; o.y = xv.y + rv.y; o.z = xv.z + rv.z; o.w = xv.w + rv.w;
        out4[i] = o;
    }
}

// ---------------------------------------------------------------------------
extern "C" void kernel_launch(void* const* d_in, const int* in_sizes, int n_in,
                              void* d_out, int out_size, void* d_ws, size_t ws_size,
                              hipStream_t stream) {
    const float* x            = (const float*)d_in[0];
    const float* w_bands      = (const float*)d_in[1];
    const float* freq_bias    = (const float*)d_in[2];
    const float* inject_gate  = (const float*)d_in[3];
    const float* blend_gate   = (const float*)d_in[4];
    const float* compress_g   = (const float*)d_in[5];
    const float* band_trust   = (const float*)d_in[6];
    const float* compress_W   = (const float*)d_in[7];
    const float* retrieve_W   = (const float*)d_in[8];
    const float* roles_r      = (const float*)d_in[9];
    const float* roles_i      = (const float*)d_in[10];
    const float* vmem_r       = (const float*)d_in[11];
    const float* vmem_i       = (const float*)d_in[12];
    const float* cheby        = (const float*)d_in[13];
    const float* lanczos      = (const float*)d_in[14];
    float* out = (float*)d_out;

    char* ws = (char*)d_ws;
    bf16_t* y_ws     = (bf16_t*)ws;                                 // 16 MiB
    float*  retr_pre = (float*)(ws + ((size_t)16 << 20));           //  1 MiB
    float*  retr_g   = (float*)(ws + ((size_t)17 << 20));           //  1 MiB

    const dim3 blk(256);
    k1_cheby_project<<<dim3(4, NWIN), blk, 0, stream>>>(x, cheby, y_ws);
    k2_compress_bind<<<dim3(8, NWIN), blk, 0, stream>>>(
        y_ws, compress_W, compress_g, w_bands, freq_bias, lanczos, band_trust,
        roles_r, roles_i, vmem_r, vmem_i, retr_pre);
    k3_retrieve<<<dim3(8, 4), blk, 0, stream>>>(
        retr_pre, retrieve_W, inject_gate, blend_gate, retr_g);
    k4_inject<<<8192, blk, 0, stream>>>(
        reinterpret_cast<const float4*>(x), retr_g, reinterpret_cast<float4*>(out));
}
